// DsaWarmupLoss_33973191311563
// MI455X (gfx1250) — matrix operand order repaired and checked by reading, not compiled, not run on hardware
//
#include <hip/hip_runtime.h>
#include <hip/hip_bf16.h>

typedef __bf16 v16bf __attribute__((ext_vector_type(16)));
typedef __bf16 v8bf  __attribute__((ext_vector_type(8)));
typedef float  v8f   __attribute__((ext_vector_type(8)));

#define S_  1024
#define H_  64
#define D_  128
#define ID_ 64

// Process one 16x16 (q-rows x k-cols) tile: 4 WMMAs for q.kT (D=128),
// 2 WMMAs for relu(iq.ikT)*w (ID=64), then lane-local online-softmax update.
// DIAG=false tiles are fully below the causal diagonal: no mask logic at all.
template<bool DIAG>
static __device__ __forceinline__
void kl_tile(const __bf16* __restrict__ kmat, const __bf16* __restrict__ ik,
             int kb, int n, int hi,
             const v16bf aq[4], const v16bf aiq[2], const float w[8],
             float m1[8], float s1[8], float a1[8], float a2[8],
             float m2[8], float s2[8])
{
    // B operand (32x16 bf16): lane supplies column N=n; lanes 0-15 hold
    // K=k0..k0+15, lanes 16-31 K=k0+16..k0+31 -> contiguous row reads.
    const __bf16* krow  = kmat + (size_t)(kb + n) * D_  + (hi ? 16 : 0);
    const __bf16* ikrow = ik   + (size_t)(kb + n) * ID_ + (hi ? 16 : 0);

    v8f c1 = {};
#pragma unroll
    for (int c = 0; c < 4; ++c) {
        v16bf b = *(const v16bf*)(krow + 32 * c);
        c1 = __builtin_amdgcn_wmma_f32_16x16x32_bf16(
                 false, aq[c], false, b, (short)0, c1, false, false);
    }
    v8f c2 = {};
#pragma unroll
    for (int c = 0; c < 2; ++c) {
        v16bf b = *(const v16bf*)(ikrow + 32 * c);
        c2 = __builtin_amdgcn_wmma_f32_16x16x32_bf16(
                 false, aiq[c], false, b, (short)0, c2, false, false);
    }

#pragma unroll
    for (int j = 0; j < 8; ++j) {
        const float l1 = c1[j];
        const float z  = c2[j];
        const float l2 = (z > 0.0f ? z : 0.0f) * w[j];
        const bool valid = DIAG ? (n <= j + (hi << 3)) : true;

        // --- online update of p = softmax(l1) moments -------------------
        // m_new = max(m,l): exactly one of {scale, p} is exp(-|l-m|),
        // the other is 1  ->  single v_exp_f32 with -|x| src modifiers.
        {
            const float d  = l1 - m1[j];
            const bool  g  = valid && (d > 0.0f);
            const float e  = __expf(-fabsf(d));
            const float sc = g ? e : 1.0f;
            float       p  = g ? 1.0f : e;
            if (DIAG) p = valid ? p : 0.0f;       // masked: state untouched
            m1[j] = g ? l1 : m1[j];
            s1[j] = fmaf(s1[j], sc, p);
            a1[j] = fmaf(a1[j], sc, p * l1);
            a2[j] = fmaf(a2[j], sc, p * l2);
        }
        // --- online LSE of l2 -------------------------------------------
        {
            const float d  = l2 - m2[j];
            const bool  g  = valid && (d > 0.0f);
            const float e  = __expf(-fabsf(d));
            const float sc = g ? e : 1.0f;
            float       p  = g ? 1.0f : e;
            if (DIAG) p = valid ? p : 0.0f;
            m2[j] = g ? l2 : m2[j];
            s2[j] = fmaf(s2[j], sc, p);
        }
    }
}

// One wave per (16-row q-tile, head). Block = 8 waves = 8 consecutive heads
// of the same q-tile (identical k-tile stream -> L2/WGP$ reuse).
__global__ __launch_bounds__(256)
void dsa_kl_kernel(const __bf16* __restrict__ q,
                   const __bf16* __restrict__ kmat,
                   const __bf16* __restrict__ iq,
                   const __bf16* __restrict__ ik,
                   const float*  __restrict__ iw,
                   float* __restrict__ out)
{
    const int lane = threadIdx.x & 31;
    const int wave = blockIdx.x * (blockDim.x >> 5) + (threadIdx.x >> 5);
    const int h  = wave & (H_ - 1);
    const int qt = wave >> 6;          // q tile (0..63)
    const int qb = qt << 4;

    const int n    = lane & 15;
    const int hi   = lane >> 4;
    const int koff = hi << 4;

    // A operands, resident in VGPRs for the whole k loop.
    // A 16x32 as this code loads it (contiguous k): lane n -> row M=n; lanes 0-15: K=k0..k0+15;
    // lanes 16-31: K=k0+16..k0+31 (matches B).
    const __bf16* qrow = q + ((size_t)(qb + n) * H_ + h) * D_;
    v16bf aq[4];
#pragma unroll
    for (int c = 0; c < 4; ++c) {
        v8bf lo = *(const v8bf*)(qrow + 32 * c + koff);
        v8bf hv = *(const v8bf*)(qrow + 32 * c + 8 + koff);
#pragma unroll
        for (int i = 0; i < 8; ++i) { aq[c][i] = lo[i]; aq[c][8 + i] = hv[i]; }
    }
    const __bf16* iqrow = iq + ((size_t)(qb + n) * H_ + h) * ID_;
    v16bf aiq[2];
#pragma unroll
    for (int c = 0; c < 2; ++c) {
        v8bf lo = *(const v8bf*)(iqrow + 32 * c + koff);
        v8bf hv = *(const v8bf*)(iqrow + 32 * c + 8 + koff);
#pragma unroll
        for (int i = 0; i < 8; ++i) { aiq[c][i] = lo[i]; aiq[c][8 + i] = hv[i]; }
    }

    float w[8];
#pragma unroll
    for (int j = 0; j < 8; ++j)
        w[j] = iw[(size_t)(qb + j + (hi << 3)) * H_ + h];

    // Lane-local online state; -1e30 bootstrap (first valid element sets
    // scale = exp(-huge) = 0, p = 1 -> clean initialization, no NaNs).
    float m1[8], s1[8], a1[8], a2[8], m2[8], s2[8];
#pragma unroll
    for (int j = 0; j < 8; ++j) {
        m1[j] = -1e30f; s1[j] = 0.f; a1[j] = 0.f; a2[j] = 0.f;
        m2[j] = -1e30f; s2[j] = 0.f;
    }

    // Hot loop: strictly-below-diagonal tiles, zero mask logic.
    for (int kt = 0; kt < qt; ++kt) {
        const int kb = kt << 4;
        // prefetch next tile's rows
        __builtin_prefetch(kmat + (size_t)(kb + 16 + n) * D_,  0, 3);
        __builtin_prefetch(ik   + (size_t)(kb + 16 + n) * ID_, 0, 3);
        kl_tile<false>(kmat, ik, kb, n, hi, aq, aiq, w, m1, s1, a1, a2, m2, s2);
    }
    // Diagonal tile with causal mask.
    kl_tile<true>(kmat, ik, qb, n, hi, aq, aiq, w, m1, s1, a1, a2, m2, s2);

    // Merge the 16 lane-states per row (xor 8,4,2,1 stays inside 16-groups).
#pragma unroll
    for (int off = 8; off >= 1; off >>= 1) {
#pragma unroll
        for (int j = 0; j < 8; ++j) {
            const float m1o = __shfl_xor(m1[j], off);
            const float s1o = __shfl_xor(s1[j], off);
            const float a1o = __shfl_xor(a1[j], off);
            const float a2o = __shfl_xor(a2[j], off);
            const float m2o = __shfl_xor(m2[j], off);
            const float s2o = __shfl_xor(s2[j], off);
            {   // one exp per pairwise merge
                const float d   = m1o - m1[j];
                const bool  g   = d > 0.0f;
                const float e   = __expf(-fabsf(d));
                const float scA = g ? e : 1.0f;
                const float scB = g ? 1.0f : e;
                m1[j] = g ? m1o : m1[j];
                s1[j] = fmaf(s1[j], scA, s1o * scB);
                a1[j] = fmaf(a1[j], scA, a1o * scB);
                a2[j] = fmaf(a2[j], scA, a2o * scB);
            }
            {
                const float d   = m2o - m2[j];
                const bool  g   = d > 0.0f;
                const float e   = __expf(-fabsf(d));
                const float scA = g ? e : 1.0f;
                const float scB = g ? 1.0f : e;
                m2[j] = g ? m2o : m2[j];
                s2[j] = fmaf(s2[j], scA, s2o * scB);
            }
        }
    }

    // KL = E_p[l1] - E_p[l2] - LSE1 + LSE2
    if (n == 0) {
#pragma unroll
        for (int j = 0; j < 8; ++j) {
            const int m = j + (hi << 3);
            const float kl = (a1[j] - a2[j]) / s1[j]
                           - (m1[j] + __logf(s1[j]))
                           + (m2[j] + __logf(s2[j]));
            out[(size_t)(qb + m) * H_ + h] = kl;
        }
    }
}

extern "C" void kernel_launch(void* const* d_in, const int* in_sizes, int n_in,
                              void* d_out, int out_size, void* d_ws, size_t ws_size,
                              hipStream_t stream) {
    (void)in_sizes; (void)n_in; (void)out_size; (void)d_ws; (void)ws_size;
    const __bf16* q  = (const __bf16*)d_in[0];
    const __bf16* k  = (const __bf16*)d_in[1];
    const __bf16* iq = (const __bf16*)d_in[2];
    const __bf16* ik = (const __bf16*)d_in[3];
    const float*  iw = (const float*)d_in[4];
    float* out = (float*)d_out;

    const int blocks = (S_ / 16) * H_ / 8;   // 512 blocks x 8 waves = 4096 waves
    dsa_kl_kernel<<<blocks, 256, 0, stream>>>(q, k, iq, ik, iw, out);
}
